// GATClassifier_91027536871850
// MI455X (gfx1250) — matrix operand order repaired and checked by reading, not compiled, not run on hardware
//
#include <hip/hip_runtime.h>
#include <hip/hip_bf16.h>
#include <math.h>

typedef __attribute__((ext_vector_type(16))) _Float16 v16h;
typedef __attribute__((ext_vector_type(8)))  _Float16 v8h;
typedef __attribute__((ext_vector_type(8)))  float    v8f;

#define NEG_SLOPE 0.2f

// ---------------------------------------------------------------------------
// helpers
// ---------------------------------------------------------------------------
__device__ inline void atomicMaxF32(float* addr, float val) {
    unsigned int* ua = (unsigned int*)addr;
    unsigned int old = *ua;
    while (true) {
        float cur = __uint_as_float(old);
        if (cur >= val) break;
        unsigned int assumed = old;
        old = atomicCAS(ua, assumed, __float_as_uint(val));
        if (old == assumed) break;
    }
}

// ---------------------------------------------------------------------------
// elementwise conversions / fills
// ---------------------------------------------------------------------------
__global__ void k_f32_to_f16(const float* __restrict__ in, _Float16* __restrict__ out,
                             size_t n) {
    size_t t = (size_t)blockIdx.x * blockDim.x + threadIdx.x;
    if (t < n) out[t] = (_Float16)in[t];
}

// in: [K][Nn] f32 row-major -> out: [Nn][K] f16 (transposed)
__global__ void k_transpose_f32_to_f16(const float* __restrict__ in,
                                       _Float16* __restrict__ out, int K, int Nn) {
    int t = blockIdx.x * blockDim.x + threadIdx.x;
    if (t >= K * Nn) return;
    int k = t / Nn, nn = t % Nn;
    out[(size_t)nn * K + k] = (_Float16)in[t];
}

__global__ void k_fill_f32(float* __restrict__ p, size_t n, float val) {
    size_t t = (size_t)blockIdx.x * blockDim.x + threadIdx.x;
    if (t < n) p[t] = val;
}

// out[n*C + c] = bias[c]
__global__ void k_init_bias(float* __restrict__ out, const float* __restrict__ bias,
                            size_t n, int C) {
    size_t t = (size_t)blockIdx.x * blockDim.x + threadIdx.x;
    if (t < n) out[t] = bias[t % C];
}

// out_f16 = elu(in + bias[c])   (layer-1 post-processing feeding GEMM2)
__global__ void k_bias_elu_to_f16(const float* __restrict__ in,
                                  const float* __restrict__ bias,
                                  _Float16* __restrict__ out, size_t n, int C) {
    size_t t = (size_t)blockIdx.x * blockDim.x + threadIdx.x;
    if (t >= n) return;
    float v = in[t] + bias[t % C];
    v = v > 0.f ? v : (expf(v) - 1.f);
    out[t] = (_Float16)v;
}

// ---------------------------------------------------------------------------
// WMMA GEMM:  C[M,N] = A[M,K] * B[K,N],  A row-major f16, BT = B^T [N,K] f16.
// Each wave32 computes a 16 x (16*NT) strip: the A fragment is loaded once per
// 32-wide K step and reused across NT WMMAs (raises arithmetic intensity; A is
// the 50000-row matrix, so this cuts its cache traffic NT-fold).
// Lane layouts per CDNA5 ISA 7.12.2.
// ---------------------------------------------------------------------------
template <int NT>
__global__ __launch_bounds__(256)
void k_wmma_gemm(const _Float16* __restrict__ A, const _Float16* __restrict__ BT,
                 float* __restrict__ C, int M, int N, int K) {
    int wave = (int)((blockIdx.x * blockDim.x + threadIdx.x) >> 5);
    int lane = threadIdx.x & 31;
    int groupsN = (N >> 4) / NT;       // strips of NT tiles along N
    int tm = wave / groupsN;
    int tg = wave % groupsN;
    if (tm * 16 >= M) return;

    int  sub = lane & 15;
    bool hi  = lane >= 16;

    const _Float16* arow = A + (size_t)(tm * 16 + sub) * K;
    const _Float16* bcol[NT];
#pragma unroll
    for (int j = 0; j < NT; ++j)
        bcol[j] = BT + (size_t)((tg * NT + j) * 16 + sub) * K;

    v8f c[NT];
#pragma unroll
    for (int j = 0; j < NT; ++j) c[j] = (v8f){};

    for (int kk = 0; kk < K; kk += 32) {
        union { v16h v; v8h h[2]; } a;
        // A 16x32 f16: lanes 0-15 take K {kk..kk+7, kk+16..kk+23},
        //              lanes 16-31 take K {kk+8..kk+15, kk+24..kk+31}
        a.h[0] = *(const v8h*)(arow + kk + (hi ? 8 : 0));
        a.h[1] = *(const v8h*)(arow + kk + 16 + (hi ? 8 : 0));
#pragma unroll
        for (int j = 0; j < NT; ++j) {
            union { v16h v; v8h h[2]; } b;
            // B 32x16 f16 (column = lane%16): lanes 0-15 K {kk..kk+7, kk+16..kk+23},
            //                                 lanes 16-31 K {kk+8..kk+15, kk+24..kk+31}
            b.h[0] = *(const v8h*)(bcol[j] + kk + (hi ? 8 : 0));
            b.h[1] = *(const v8h*)(bcol[j] + kk + 16 + (hi ? 8 : 0));
            c[j] = __builtin_amdgcn_wmma_f32_16x16x32_f16(
                       /*neg_a=*/false, a.v, /*neg_b=*/false, b.v,
                       /*c_mod=*/(short)0, c[j], /*reuse_a=*/false, /*reuse_b=*/false);
        }
    }

    // C/D layout: VGPR r -> row (r + hi*8), col = lane%16
#pragma unroll
    for (int j = 0; j < NT; ++j) {
        float* crow = C + (size_t)(tm * 16 + (hi ? 8 : 0)) * N + (tg * NT + j) * 16 + sub;
#pragma unroll
        for (int r = 0; r < 8; ++r) crow[(size_t)r * N] = c[j][r];
    }
}

// ---------------------------------------------------------------------------
// attention coefficients: as[n,h] = <feat[n,h,:], a_src[h,:]>, same for ad.
// One wave per (node, head).
// ---------------------------------------------------------------------------
__global__ __launch_bounds__(256)
void k_alpha(const float* __restrict__ feat, const float* __restrict__ a_src,
             const float* __restrict__ a_dst, float* __restrict__ as,
             float* __restrict__ ad, int Nn, int H, int D) {
    int wave = (int)((blockIdx.x * blockDim.x + threadIdx.x) >> 5);
    int lane = threadIdx.x & 31;
    if (wave >= Nn * H) return;
    int hh = wave % H;
    const float* hp = feat + (size_t)wave * D;   // feat is [N, H, D] contiguous
    float s = 0.f, d = 0.f;
    for (int i = lane; i < D; i += 32) {
        float v = hp[i];
        s += v * a_src[hh * D + i];
        d += v * a_dst[hh * D + i];
    }
#pragma unroll
    for (int off = 16; off; off >>= 1) {
        s += __shfl_xor(s, off);
        d += __shfl_xor(d, off);
    }
    if (lane == 0) { as[wave] = s; ad[wave] = d; }
}

// ---------------------------------------------------------------------------
// edge passes. Edges e < E come from edge_index; e >= E are self-loops.
// ---------------------------------------------------------------------------
__device__ inline void edge_ends(const int* __restrict__ ei, int e, int E,
                                 int* s, int* d) {
    if (e < E) { *s = ei[e]; *d = ei[E + e]; }
    else       { *s = *d = e - E; }
}

__global__ void k_edge_max(const int* __restrict__ ei, const float* __restrict__ as,
                           const float* __restrict__ ad, float* __restrict__ emax,
                           int E, int Etot, int H) {
    int t = blockIdx.x * blockDim.x + threadIdx.x;
    if (t >= Etot * H) return;
    int e = t / H, hh = t % H;
    int s, d;
    edge_ends(ei, e, E, &s, &d);
    float v = as[s * H + hh] + ad[d * H + hh];
    v = v > 0.f ? v : NEG_SLOPE * v;
    atomicMaxF32(&emax[d * H + hh], v);
}

__global__ void k_edge_expsum(const int* __restrict__ ei, const float* __restrict__ as,
                              const float* __restrict__ ad, const float* __restrict__ emax,
                              float* __restrict__ ex, float* __restrict__ den,
                              int E, int Etot, int H) {
    int t = blockIdx.x * blockDim.x + threadIdx.x;
    if (t >= Etot * H) return;
    int e = t / H, hh = t % H;
    int s, d;
    edge_ends(ei, e, E, &s, &d);
    float v = as[s * H + hh] + ad[d * H + hh];
    v = v > 0.f ? v : NEG_SLOPE * v;
    float ev = expf(v - emax[d * H + hh]);
    ex[(size_t)e * H + hh] = ev;
    atomicAdd(&den[d * H + hh], ev);
}

// one wave per (edge, head); lanes stride the feature dim
__global__ __launch_bounds__(256)
void k_edge_agg(const int* __restrict__ ei, const float* __restrict__ feat,
                const float* __restrict__ ex, const float* __restrict__ den,
                float* __restrict__ out, int E, int Etot, int H, int D) {
    int wave = (int)((blockIdx.x * blockDim.x + threadIdx.x) >> 5);
    int lane = threadIdx.x & 31;
    if (wave >= Etot * H) return;
    int e = wave / H, hh = wave % H;
    int s, d;
    edge_ends(ei, e, E, &s, &d);
    float alpha = ex[(size_t)e * H + hh] / (den[d * H + hh] + 1e-16f);
    const float* hp = feat + ((size_t)s * H + hh) * D;
    float*       op = out  + ((size_t)d * H + hh) * D;
    for (int i = lane; i < D; i += 32)
        atomicAdd(&op[i], hp[i] * alpha);
}

// ---------------------------------------------------------------------------
// launch
// ---------------------------------------------------------------------------
static inline int cdiv(long long a, long long b) { return (int)((a + b - 1) / b); }

extern "C" void kernel_launch(void* const* d_in, const int* in_sizes, int n_in,
                              void* d_out, int out_size, void* d_ws, size_t ws_size,
                              hipStream_t stream) {
    const float* x   = (const float*)d_in[0];
    const int*   ei  = (const int*)  d_in[1];   // [2, E]
    // d_in[2] = edge_attr (ignored by the reference)
    const float* W1  = (const float*)d_in[3];   // [256, 512]
    const float* a1s = (const float*)d_in[4];   // [4, 128]
    const float* a1d = (const float*)d_in[5];
    const float* b1  = (const float*)d_in[6];   // [512]
    const float* W2  = (const float*)d_in[7];   // [512, 64]
    const float* a2s = (const float*)d_in[8];   // [1, 64]
    const float* a2d = (const float*)d_in[9];
    const float* b2  = (const float*)d_in[10];  // [64]
    float* out = (float*)d_out;

    const int IN = 256, C1 = 512, H1h = 4, D1d = 128, C2 = 64;
    const int Nn   = in_sizes[0] / IN;
    const int E    = in_sizes[1] / 2;
    const int Etot = E + Nn;

    // ---- workspace layout ----
    char*  ws  = (char*)d_ws;
    size_t off = 0;
    auto alloc = [&](size_t bytes) { size_t o = off; off = (off + bytes + 255) & ~(size_t)255; return o; };
    size_t o_xh   = alloc((size_t)Nn * IN * 2);      // x in f16
    size_t o_W1t  = alloc((size_t)C1 * IN * 2);      // W1^T f16 [512,256]
    size_t o_h1   = alloc((size_t)Nn * C1 * 4);      // GEMM1 out f32 (later reused as f16 elu(out1))
    size_t o_as1  = alloc((size_t)Nn * H1h * 4);
    size_t o_ad1  = alloc((size_t)Nn * H1h * 4);
    size_t o_em1  = alloc((size_t)Nn * H1h * 4);
    size_t o_dn1  = alloc((size_t)Nn * H1h * 4);
    size_t o_ex1  = alloc((size_t)Etot * H1h * 4);
    size_t o_out1 = alloc((size_t)Nn * C1 * 4);
    size_t o_W2t  = alloc((size_t)C2 * C1 * 2);      // W2^T f16 [64,512]
    size_t o_h2   = alloc((size_t)Nn * C2 * 4);
    size_t o_as2  = alloc((size_t)Nn * 4);
    size_t o_ad2  = alloc((size_t)Nn * 4);
    size_t o_em2  = alloc((size_t)Nn * 4);
    size_t o_dn2  = alloc((size_t)Nn * 4);
    size_t o_ex2  = alloc((size_t)Etot * 4);
    (void)ws_size;

    _Float16* xh   = (_Float16*)(ws + o_xh);
    _Float16* W1t  = (_Float16*)(ws + o_W1t);
    float*    h1   = (float*)   (ws + o_h1);
    _Float16* h1h  = (_Float16*)(ws + o_h1);     // reuse: h1 dead after aggregation
    float*    as1  = (float*)(ws + o_as1);
    float*    ad1  = (float*)(ws + o_ad1);
    float*    em1  = (float*)(ws + o_em1);
    float*    dn1  = (float*)(ws + o_dn1);
    float*    ex1  = (float*)(ws + o_ex1);
    float*    out1 = (float*)(ws + o_out1);
    _Float16* W2t  = (_Float16*)(ws + o_W2t);
    float*    h2   = (float*)(ws + o_h2);
    float*    as2  = (float*)(ws + o_as2);
    float*    ad2  = (float*)(ws + o_ad2);
    float*    em2  = (float*)(ws + o_em2);
    float*    dn2  = (float*)(ws + o_dn2);
    float*    ex2  = (float*)(ws + o_ex2);

    const int B = 256;

    // ---- layer 1 ----
    k_f32_to_f16<<<cdiv((long long)Nn * IN, B), B, 0, stream>>>(x, xh, (size_t)Nn * IN);
    k_transpose_f32_to_f16<<<cdiv((long long)IN * C1, B), B, 0, stream>>>(W1, W1t, IN, C1);

    {   // GEMM1: [N,256] x [256,512], 16x64 strip per wave
        long long waves = (long long)(Nn / 16) * ((C1 / 16) / 4);
        k_wmma_gemm<4><<<cdiv(waves * 32, B), B, 0, stream>>>(xh, W1t, h1, Nn, C1, IN);
    }

    k_alpha<<<cdiv((long long)Nn * H1h * 32, B), B, 0, stream>>>(h1, a1s, a1d, as1, ad1, Nn, H1h, D1d);

    k_fill_f32<<<cdiv((long long)Nn * H1h, B), B, 0, stream>>>(em1, (size_t)Nn * H1h, -3.0e38f);
    k_fill_f32<<<cdiv((long long)Nn * H1h, B), B, 0, stream>>>(dn1, (size_t)Nn * H1h, 0.f);
    k_fill_f32<<<cdiv((long long)Nn * C1, B), B, 0, stream>>>(out1, (size_t)Nn * C1, 0.f);

    k_edge_max<<<cdiv((long long)Etot * H1h, B), B, 0, stream>>>(ei, as1, ad1, em1, E, Etot, H1h);
    k_edge_expsum<<<cdiv((long long)Etot * H1h, B), B, 0, stream>>>(ei, as1, ad1, em1, ex1, dn1, E, Etot, H1h);
    k_edge_agg<<<cdiv((long long)Etot * H1h * 32, B), B, 0, stream>>>(ei, h1, ex1, dn1, out1, E, Etot, H1h, D1d);

    // bias + ELU, convert to f16 (into old h1 storage)
    k_bias_elu_to_f16<<<cdiv((long long)Nn * C1, B), B, 0, stream>>>(out1, b1, h1h, (size_t)Nn * C1, C1);

    // ---- layer 2 ----
    k_transpose_f32_to_f16<<<cdiv((long long)C1 * C2, B), B, 0, stream>>>(W2, W2t, C1, C2);
    {   // GEMM2: [N,512] x [512,64]; one wave covers the whole N extent
        long long waves = (long long)(Nn / 16) * ((C2 / 16) / 4);
        k_wmma_gemm<4><<<cdiv(waves * 32, B), B, 0, stream>>>(h1h, W2t, h2, Nn, C2, C1);
    }

    k_alpha<<<cdiv((long long)Nn * 32, B), B, 0, stream>>>(h2, a2s, a2d, as2, ad2, Nn, 1, C2);

    k_fill_f32<<<cdiv((long long)Nn, B), B, 0, stream>>>(em2, (size_t)Nn, -3.0e38f);
    k_fill_f32<<<cdiv((long long)Nn, B), B, 0, stream>>>(dn2, (size_t)Nn, 0.f);
    k_init_bias<<<cdiv((long long)Nn * C2, B), B, 0, stream>>>(out, b2, (size_t)Nn * C2, C2);

    k_edge_max<<<cdiv((long long)Etot, B), B, 0, stream>>>(ei, as2, ad2, em2, E, Etot, 1);
    k_edge_expsum<<<cdiv((long long)Etot, B), B, 0, stream>>>(ei, as2, ad2, em2, ex2, dn2, E, Etot, 1);
    k_edge_agg<<<cdiv((long long)Etot * 32, B), B, 0, stream>>>(ei, h2, ex2, dn2, out, E, Etot, 1, C2);
}
